// SwinTransformerBlock_34969623724454
// MI455X (gfx1250) — matrix operand order repaired and checked by reading, not compiled, not run on hardware
//
#include <hip/hip_runtime.h>

// ---------------------------------------------------------------------------
// Swin Transformer block for gfx1250 (CDNA5), bf16 WMMA with f32 accumulate.
// Data staging uses CDNA5 async-to-LDS + Tensor Data Mover (TDM).
// B=16, H=W=64, C=256, WS=8, SHIFT=4, HEADS=8, HEAD_DIM=32, N=64, HID=1024
// ---------------------------------------------------------------------------

typedef __attribute__((ext_vector_type(16))) __bf16   v16bf;
typedef __attribute__((ext_vector_type(8)))  float    v8f;
typedef __attribute__((ext_vector_type(4)))  __bf16   bf4;
typedef __attribute__((ext_vector_type(4)))  unsigned u32x4;
typedef __attribute__((ext_vector_type(8)))  int      i32x8;
typedef __attribute__((ext_vector_type(4)))  int      i32x4;

#define HAVE_TDM __has_builtin(__builtin_amdgcn_tensor_load_to_lds)

static __device__ inline v8f wmma_bf16(v16bf a, v16bf b, v8f c) {
  return __builtin_amdgcn_wmma_f32_16x16x32_bf16(false, a, false, b, (short)0, c,
                                                 false, false);
}

// Async memory->LDS copy of 16 bytes (per-lane), tracked by ASYNCcnt.
static __device__ inline void async_b128(void* lds, const void* gptr) {
  unsigned loff = (unsigned)(uintptr_t)lds;               // LDS offset (addr[31:0])
  unsigned long long ga = (unsigned long long)(uintptr_t)gptr;
  asm volatile("global_load_async_to_lds_b128 %0, %1, off"
               :: "v"(loff), "v"(ga) : "memory");
}
static __device__ inline void wait_async0() {
  asm volatile("s_wait_asynccnt 0x0" ::: "memory");
}

#if HAVE_TDM
// TDM: DMA a 2D tile (tile0 x tile1 bf16 elements, row stride stride0
// elements) from global memory into LDS (contiguous row-major).
static __device__ inline void tdm_load_2d(void* lds, const void* gptr,
                                          int tile0, int tile1,
                                          long long stride0) {
  unsigned long long ga = (unsigned long long)(uintptr_t)gptr;
  unsigned long long s0 = (unsigned long long)stride0;
  u32x4 g0;
  g0[0] = 1u;                                             // count=1 valid D#
  g0[1] = (unsigned)(uintptr_t)lds;                       // lds_addr
  g0[2] = (unsigned)(ga & 0xffffffffu);                   // global_addr[31:0]
  g0[3] = (unsigned)((ga >> 32) & 0x01ffffffu) | 0x80000000u;  // addr hi | type=2
  i32x8 g1;
  g1[0] = (int)(1u << 16);                                // data_size = 2 bytes
  g1[1] = (int)0xffff0000u;                               // tensor_dim0 lo16 (2^31-1)
  g1[2] = (int)0xffff7fffu;                               // td0 hi16 | td1 lo16
  g1[3] = (int)(0x7fffu | ((unsigned)tile0 << 16));       // td1 hi16 | tile_dim0
  g1[4] = (int)(unsigned)tile1;                           // tile_dim1 (tile_dim2=0)
  g1[5] = (int)(unsigned)(s0 & 0xffffffffu);              // tensor_dim0_stride lo32
  g1[6] = (int)(unsigned)((s0 >> 32) & 0xffffu);          // stride0 hi16 | stride1 lo
  g1[7] = 0;
  i32x4 gz = {0, 0, 0, 0};
#if __clang_major__ >= 23
  i32x8 z8 = {0, 0, 0, 0, 0, 0, 0, 0};
  __builtin_amdgcn_tensor_load_to_lds(g0, g1, gz, gz, z8, 0);
#else
  __builtin_amdgcn_tensor_load_to_lds(g0, g1, gz, gz, 0);
#endif
}
#endif

// A fragment 16x32 (MxK) bf16, ISA 7.12.2 layout:
//   m = lane&15 (both halves), k = {0..7|8..15|16..23|24..31} split per half.
static __device__ inline v16bf load_frag_a(const __bf16* src, int strideM,
                                           int strideK, int lane) {
  int m  = lane & 15;
  int hl = lane >> 4;
  v16bf f;
#pragma unroll
  for (int g = 0; g < 8; ++g) {
    int kb = ((g & 4) << 2) + 2 * (g & 3) + 8 * hl;
    f[2 * g]     = src[m * strideM + kb * strideK];
    f[2 * g + 1] = src[m * strideM + (kb + 1) * strideK];
  }
  return f;
}

// B fragment 32x16 (KxN) bf16: n = lane&15, k = 8*half + e (e<8), 16 + 8*half + (e-8) (e>=8).
static __device__ inline v16bf load_frag_b(const __bf16* src, int strideK,
                                           int strideN, int lane) {
  int n  = lane & 15;
  int hl = lane >> 4;
  v16bf f;
#pragma unroll
  for (int e = 0; e < 16; ++e) {
    int k = ((e & 8) << 1) + 8 * hl + (e & 7);
    f[e]  = src[k * strideK + n * strideN];
  }
  return f;
}

static __device__ inline float wred_sum(float v) {
#pragma unroll
  for (int m = 16; m >= 1; m >>= 1) v += __shfl_xor(v, m, 32);
  return v;
}

// region id for the shifted-window attention mask (slices at 56 and 60).
static __device__ inline int region_of(int wloc, int tok) {
  int yy = ((wloc >> 3) << 3) + (tok >> 3);
  int xx = ((wloc & 7) << 3) + (tok & 7);
  int rh = (yy < 56) ? 0 : ((yy < 60) ? 1 : 2);
  int rw = (xx < 56) ? 0 : ((xx < 60) ? 1 : 2);
  return rh * 3 + rw;
}

// ---------------------------------------------------------------------------
// Prep: fp32 weights -> bf16, and gather bias[h][i][j] = table[rel_index[i,j], h]
// ---------------------------------------------------------------------------
__global__ __launch_bounds__(256) void prep_kernel(
    const float* __restrict__ qkv_w, const float* __restrict__ proj_w,
    const float* __restrict__ w1, const float* __restrict__ w2,
    const float* __restrict__ table, const int* __restrict__ relidx,
    __bf16* __restrict__ wq, __bf16* __restrict__ wp, __bf16* __restrict__ w1b,
    __bf16* __restrict__ w2b, float* __restrict__ biasT) {
  int i = blockIdx.x * 256 + threadIdx.x;
  if (i < 196608) {
    wq[i] = (__bf16)qkv_w[i];
  } else if (i < 262144) {
    wp[i - 196608] = (__bf16)proj_w[i - 196608];
  } else if (i < 524288) {
    w1b[i - 262144] = (__bf16)w1[i - 262144];
  } else if (i < 786432) {
    w2b[i - 524288] = (__bf16)w2[i - 524288];
  } else if (i < 819200) {
    int t = i - 786432;
    int h = t >> 12;
    int ij = t & 4095;
    biasT[t] = table[relidx[ij] * 8 + h];
  }
}

// ---------------------------------------------------------------------------
// LayerNorm over C=256. One wave per row (8 rows / block of 256 threads).
// ---------------------------------------------------------------------------
template <bool SHIFTED>
__global__ __launch_bounds__(256) void ln_kernel(
    const float* __restrict__ src, const float* __restrict__ gamma,
    const float* __restrict__ beta, __bf16* __restrict__ dst) {
  int wave = threadIdx.x >> 5, lane = threadIdx.x & 31;
  int r = blockIdx.x * 8 + wave;
  int td;
  if (SHIFTED) {
    int wI = r >> 6, tok = r & 63;
    int bI = wI >> 6, wl = wI & 63;
    int ys = ((wl >> 3) << 3) + (tok >> 3);
    int xs = ((wl & 7) << 3) + (tok & 7);
    int y = (ys + 4) & 63, x = (xs + 4) & 63;  // roll(-SHIFT)
    td = bI * 4096 + y * 64 + x;
  } else {
    td = r;
  }
  const float* row = src + (size_t)td * 256;
  float4 a0 = ((const float4*)row)[lane];
  float4 a1 = ((const float4*)row)[lane + 32];
  float ps = a0.x + a0.y + a0.z + a0.w + a1.x + a1.y + a1.z + a1.w;
  float mu = wred_sum(ps) * (1.0f / 256.0f);
  float d0 = a0.x - mu, d1 = a0.y - mu, d2 = a0.z - mu, d3 = a0.w - mu;
  float d4 = a1.x - mu, d5 = a1.y - mu, d6 = a1.z - mu, d7 = a1.w - mu;
  float pv = d0 * d0 + d1 * d1 + d2 * d2 + d3 * d3 + d4 * d4 + d5 * d5 +
             d6 * d6 + d7 * d7;
  float var = wred_sum(pv) * (1.0f / 256.0f);
  float rstd = rsqrtf(var + 1e-5f);
  float4 g0 = ((const float4*)gamma)[lane];
  float4 g1 = ((const float4*)gamma)[lane + 32];
  float4 b0 = ((const float4*)beta)[lane];
  float4 b1 = ((const float4*)beta)[lane + 32];
  bf4 o0, o1;
  o0[0] = (__bf16)(d0 * rstd * g0.x + b0.x);
  o0[1] = (__bf16)(d1 * rstd * g0.y + b0.y);
  o0[2] = (__bf16)(d2 * rstd * g0.z + b0.z);
  o0[3] = (__bf16)(d3 * rstd * g0.w + b0.w);
  o1[0] = (__bf16)(d4 * rstd * g1.x + b1.x);
  o1[1] = (__bf16)(d5 * rstd * g1.y + b1.y);
  o1[2] = (__bf16)(d6 * rstd * g1.z + b1.z);
  o1[3] = (__bf16)(d7 * rstd * g1.w + b1.w);
  bf4* drow = (bf4*)(dst + (size_t)r * 256);
  drow[lane]      = o0;
  drow[lane + 32] = o1;
}

// ---------------------------------------------------------------------------
// Generic 64x256-tile GEMM, bf16 in / f32 accumulate, 8 waves per block.
// A tile staged via async-to-LDS; B tile via one TDM descriptor per chunk.
// ---------------------------------------------------------------------------
template <int EPI, int KTOT, int NLD>
__global__ __launch_bounds__(256) void gemm_kernel(
    const __bf16* __restrict__ A, int lda, const __bf16* __restrict__ Bw,
    const float* __restrict__ bias, __bf16* __restrict__ outb,
    float* __restrict__ outf, const float* __restrict__ resid) {
  __shared__ __bf16 As[64 * 64];    // 8 KB
  __shared__ __bf16 Bs[64 * 256];   // 32 KB
  int tid = threadIdx.x, lane = tid & 31, wave = tid >> 5;
  int row0 = blockIdx.x * 64, n0 = blockIdx.y * 256;

  v8f zero = {0.f, 0.f, 0.f, 0.f, 0.f, 0.f, 0.f, 0.f};
  v8f acc[4][2];
#pragma unroll
  for (int i = 0; i < 4; ++i)
#pragma unroll
    for (int j = 0; j < 2; ++j) acc[i][j] = zero;

  for (int kc = 0; kc < KTOT; kc += 64) {
#if HAVE_TDM
    if (wave == 0)  // whole 64x256 B tile in one TDM transfer
      tdm_load_2d(Bs, Bw + (size_t)kc * NLD + n0, 256, 64, (long long)NLD);
#else
#pragma unroll
    for (int q = 0; q < 8; ++q) {
      int idx = q * 256 + tid;
      int br = idx >> 5, sg = idx & 31;
      async_b128(Bs + br * 256 + sg * 8,
                 Bw + (size_t)(kc + br) * NLD + n0 + sg * 8);
    }
#endif
#pragma unroll
    for (int q = 0; q < 2; ++q) {  // A tile: 64x64 bf16, async to LDS
      int idx = q * 256 + tid;
      int ar = idx >> 3, sg = idx & 7;
      async_b128(As + ar * 64 + sg * 8,
                 A + (size_t)(row0 + ar) * lda + kc + sg * 8);
    }
#if HAVE_TDM
    if (wave == 0) __builtin_amdgcn_s_wait_tensorcnt(0);
#endif
    wait_async0();
    __syncthreads();
#pragma unroll
    for (int kk = 0; kk < 64; kk += 32) {
#pragma unroll
      for (int rt = 0; rt < 4; ++rt) {
        v16bf a = load_frag_a(As + rt * 16 * 64 + kk, 64, 1, lane);
#pragma unroll
        for (int ct = 0; ct < 2; ++ct) {
          v16bf b = load_frag_b(Bs + kk * 256 + wave * 32 + ct * 16, 256, 1, lane);
          acc[rt][ct] = wmma_bf16(a, b, acc[rt][ct]);
        }
      }
    }
    __syncthreads();
  }

  int hl = lane >> 4, ll = lane & 15;
#pragma unroll
  for (int ct = 0; ct < 2; ++ct) {
    int gcol = n0 + wave * 32 + ct * 16 + ll;
    float bc = bias[gcol];
#pragma unroll
    for (int rt = 0; rt < 4; ++rt) {
#pragma unroll
      for (int r = 0; r < 8; ++r) {
        int grow = row0 + rt * 16 + r + 8 * hl;
        float v = acc[rt][ct][r] + bc;
        if (EPI == 0) {
          outb[(size_t)grow * NLD + gcol] = (__bf16)v;
        } else if (EPI == 1) {
          int wI = grow >> 6, tok = grow & 63;
          int bI = wI >> 6, wl = wI & 63;
          int ys = ((wl >> 3) << 3) + (tok >> 3);
          int xs = ((wl & 7) << 3) + (tok & 7);
          int y = (ys + 4) & 63, x = (xs + 4) & 63;  // window reverse + roll(+4)
          size_t td = (size_t)bI * 4096 + y * 64 + x;
          outf[td * 256 + gcol] = resid[td * 256 + gcol] + v;
        } else if (EPI == 2) {
          float g = 0.5f * v *
                    (1.0f + tanhf(0.7978845608028654f * (v + 0.044715f * v * v * v)));
          outb[(size_t)grow * NLD + gcol] = (__bf16)g;
        } else {
          outf[(size_t)grow * 256 + gcol] = resid[(size_t)grow * 256 + gcol] + v;
        }
      }
    }
  }
}

// ---------------------------------------------------------------------------
// Window attention: one block per window (of 1024), one wave per head.
// k/v staged by TDM as [tok][8*32] tiles (stride-768 rows -> contiguous LDS).
// ---------------------------------------------------------------------------
__global__ __launch_bounds__(256) void attn_kernel(
    const __bf16* __restrict__ qkv, const float* __restrict__ biasT,
    __bf16* __restrict__ out) {
  __shared__ __bf16 sk[64 * 256];      // 32 KB  k: [tok][h*32+d]
  __shared__ __bf16 sv[64 * 256];      // 32 KB  v: [tok][h*32+d]
  __shared__ __bf16 pb[8 * 16 * 64];   // 16 KB  per-wave P strip
  int tid = threadIdx.x, lane = tid & 31, wave = tid >> 5;
  int w = blockIdx.x, wloc = w & 63;
  const __bf16* base = qkv + (size_t)w * 49152;  // [64 tok][768]

#if HAVE_TDM
  if (wave == 0) tdm_load_2d(sk, base + 256, 256, 64, 768);   // k block
  if (wave == 1) tdm_load_2d(sv, base + 512, 256, 64, 768);   // v block
  if (wave < 2) __builtin_amdgcn_s_wait_tensorcnt(0);
#else
#pragma unroll
  for (int q = 0; q < 16; ++q) {
    int idx = q * 256 + tid;                    // 0..4095
    int mat = idx >> 11, t = (idx >> 5) & 63, sg = idx & 31;
    async_b128((mat ? sv : sk) + t * 256 + sg * 8,
               base + t * 768 + 256 + mat * 256 + sg * 8);
  }
  wait_async0();
#endif
  __syncthreads();

  int h = wave, hl = lane >> 4, ll = lane & 15;
  const __bf16* kh = sk + h * 32;   // (kd, n) at kh[n*256 + kd]
  const __bf16* vh = sv + h * 32;   // (kt, d) at vh[kt*256 + d]
  __bf16* ph = pb + h * 1024;
  const float* bh = biasT + h * 4096;
  const float scale = 0.17677669529663689f;  // 1/sqrt(32)
  v8f zero = {0.f, 0.f, 0.f, 0.f, 0.f, 0.f, 0.f, 0.f};

  int regj[4];
#pragma unroll
  for (int ct = 0; ct < 4; ++ct) regj[ct] = region_of(wloc, ct * 16 + ll);

  for (int s = 0; s < 4; ++s) {
    // q rows of this strip, straight from global (read once)
    v16bf aq = load_frag_a(base + (s * 16) * 768 + h * 32, 768, 1, lane);
    v8f sc[4];
#pragma unroll
    for (int ct = 0; ct < 4; ++ct) {
      sc[ct] = zero;
      v16bf bk = load_frag_b(kh + ct * 16 * 256, 1, 256, lane);  // k^T
      sc[ct] = wmma_bf16(aq, bk, sc[ct]);
    }
    // scale + relative-position bias + shifted-window mask
#pragma unroll
    for (int r = 0; r < 8; ++r) {
      int i = s * 16 + r + 8 * hl;
      int regi = region_of(wloc, i);
#pragma unroll
      for (int ct = 0; ct < 4; ++ct) {
        int j = ct * 16 + ll;
        sc[ct][r] = sc[ct][r] * scale + bh[i * 64 + j] +
                    ((regi == regj[ct]) ? 0.0f : -100.0f);
      }
    }
    // softmax per row (64 values in 4 tiles x 16 lanes of this half)
#pragma unroll
    for (int r = 0; r < 8; ++r) {
      float m = fmaxf(fmaxf(sc[0][r], sc[1][r]), fmaxf(sc[2][r], sc[3][r]));
#pragma unroll
      for (int msk = 8; msk >= 1; msk >>= 1) m = fmaxf(m, __shfl_xor(m, msk, 32));
      float e[4], ssum = 0.f;
#pragma unroll
      for (int ct = 0; ct < 4; ++ct) {
        e[ct] = __expf(sc[ct][r] - m);
        ssum += e[ct];
      }
#pragma unroll
      for (int msk = 8; msk >= 1; msk >>= 1) ssum += __shfl_xor(ssum, msk, 32);
      float inv = 1.0f / ssum;
      int rl = r + 8 * hl;
#pragma unroll
      for (int ct = 0; ct < 4; ++ct)
        ph[rl * 64 + ct * 16 + ll] = (__bf16)(e[ct] * inv);
    }
    __syncthreads();  // C-layout -> A-layout restage of P via LDS

    // out strip = P(16x64) @ v(64x32)
#pragma unroll
    for (int dt = 0; dt < 2; ++dt) {
      v8f o = zero;
#pragma unroll
      for (int kk = 0; kk < 64; kk += 32) {
        v16bf ap = load_frag_a(ph + kk, 64, 1, lane);
        v16bf bv = load_frag_b(vh + kk * 256 + dt * 16, 256, 1, lane);
        o = wmma_bf16(ap, bv, o);
      }
#pragma unroll
      for (int r = 0; r < 8; ++r) {
        size_t grow = (size_t)w * 64 + s * 16 + r + 8 * hl;
        out[grow * 256 + h * 32 + dt * 16 + ll] = (__bf16)o[r];
      }
    }
    __syncthreads();
  }
}

// ---------------------------------------------------------------------------
// Launch
// ---------------------------------------------------------------------------
extern "C" void kernel_launch(void* const* d_in, const int* in_sizes, int n_in,
                              void* d_out, int out_size, void* d_ws,
                              size_t ws_size, hipStream_t stream) {
  const float* x      = (const float*)d_in[0];
  const float* ln1_g  = (const float*)d_in[1];
  const float* ln1_b  = (const float*)d_in[2];
  const float* qkv_w  = (const float*)d_in[3];
  const float* qkv_b  = (const float*)d_in[4];
  const float* proj_w = (const float*)d_in[5];
  const float* proj_b = (const float*)d_in[6];
  const float* rtab   = (const float*)d_in[7];
  const float* ln2_g  = (const float*)d_in[8];
  const float* ln2_b  = (const float*)d_in[9];
  const float* w1     = (const float*)d_in[10];
  const float* b1     = (const float*)d_in[11];
  const float* w2     = (const float*)d_in[12];
  const float* b2     = (const float*)d_in[13];
  const int*   ridx   = (const int*)d_in[14];
  (void)in_sizes; (void)n_in; (void)out_size; (void)ws_size;

  char* ws = (char*)d_ws;
  __bf16* wqb  = (__bf16*)(ws + 0);              //   393,216 B  qkv_w bf16
  __bf16* wpb  = (__bf16*)(ws + 393216);         //   131,072 B  proj_w bf16
  __bf16* w1b  = (__bf16*)(ws + 524288);         //   524,288 B  mlp_w1 bf16
  __bf16* w2b  = (__bf16*)(ws + 1048576);        //   524,288 B  mlp_w2 bf16
  float*  bias = (float*)(ws + 1572864);         //   131,072 B  [8][64][64]
  __bf16* xw   = (__bf16*)(ws + 1703936);        //  33,554,432 B [1024][64][256]
  __bf16* aout = xw;                             //  alias: attn out (xw dead)
  __bf16* qkvb = (__bf16*)(ws + 35258368);       // 100,663,296 B [1024][64][768]
  __bf16* m_in = qkvb;                           //  alias: LN2 out (qkv dead)
  float*  hbuf = (float*)(ws + 135921664);       //  67,108,864 B h (post-attn resid)
  __bf16* hid  = (__bf16*)(ws + 203030528);      // 134,217,728 B [65536][1024]

  prep_kernel<<<3200, 256, 0, stream>>>(qkv_w, proj_w, w1, w2, rtab, ridx, wqb,
                                        wpb, w1b, w2b, bias);
  ln_kernel<true><<<8192, 256, 0, stream>>>(x, ln1_g, ln1_b, xw);
  gemm_kernel<0, 256, 768><<<dim3(1024, 3), 256, 0, stream>>>(
      xw, 256, wqb, qkv_b, qkvb, nullptr, nullptr);
  attn_kernel<<<1024, 256, 0, stream>>>(qkvb, bias, aout);
  gemm_kernel<1, 256, 256><<<dim3(1024, 1), 256, 0, stream>>>(
      aout, 256, wpb, proj_b, nullptr, hbuf, x);
  ln_kernel<false><<<8192, 256, 0, stream>>>(hbuf, ln2_g, ln2_b, m_in);
  gemm_kernel<2, 256, 1024><<<dim3(1024, 4), 256, 0, stream>>>(
      m_in, 256, w1b, b1, hid, nullptr, nullptr);
  gemm_kernel<3, 1024, 256><<<dim3(1024, 1), 256, 0, stream>>>(
      hid, 1024, w2b, b2, nullptr, (float*)d_out, hbuf);
}
